// GLAMemoryCompressor_53687091200016
// MI455X (gfx1250) — compile-verified, matrix operand order checked by reading, not run on hardware
//
#include <hip/hip_runtime.h>
#include <hip/hip_bf16.h>
#include <math.h>

// ---------- problem constants ----------
#define Bb     2
#define Tt     2048
#define Dd     2048
#define Hh     16
#define DK     64
#define DV     128
#define CHUNK  64
#define NCHUNK (Tt / CHUNK)       // 32
#define ROWS   (Bb * Tt)          // 4096
#define NK     (Hh * DK)          // 1024
#define NV     (Hh * DV)          // 2048

typedef _Float16 v16h __attribute__((ext_vector_type(16)));
typedef _Float16 v8h  __attribute__((ext_vector_type(8)));
typedef float    v8f  __attribute__((ext_vector_type(8)));

// =====================================================================
// 1) RMSNorm + cast to f16.  One block per token row, 256 thr x 8 elems.
// =====================================================================
__global__ __launch_bounds__(256) void rmsnorm_cast_kernel(
    const float* __restrict__ x, const float* __restrict__ w,
    _Float16* __restrict__ xn)
{
    __shared__ float red[8];
    const int row  = blockIdx.x;
    const int t    = threadIdx.x;
    const float* xr = x + (size_t)row * Dd + t * 8;
    float4 a = *(const float4*)(xr);
    float4 b = *(const float4*)(xr + 4);
    float ss = a.x*a.x + a.y*a.y + a.z*a.z + a.w*a.w
             + b.x*b.x + b.y*b.y + b.z*b.z + b.w*b.w;
    #pragma unroll
    for (int off = 16; off > 0; off >>= 1) ss += __shfl_xor(ss, off, 32);
    if ((t & 31) == 0) red[t >> 5] = ss;
    __syncthreads();
    float tot = 0.f;
    #pragma unroll
    for (int i = 0; i < 8; ++i) tot += red[i];
    const float scale = rsqrtf(tot * (1.0f / (float)Dd) + 1e-5f);

    const float* wr = w + t * 8;
    float4 wa = *(const float4*)(wr);
    float4 wb = *(const float4*)(wr + 4);
    v8h h;
    h[0] = (_Float16)(a.x * scale * wa.x);
    h[1] = (_Float16)(a.y * scale * wa.y);
    h[2] = (_Float16)(a.z * scale * wa.z);
    h[3] = (_Float16)(a.w * scale * wa.w);
    h[4] = (_Float16)(b.x * scale * wb.x);
    h[5] = (_Float16)(b.y * scale * wb.y);
    h[6] = (_Float16)(b.z * scale * wb.z);
    h[7] = (_Float16)(b.w * scale * wb.w);
    *(v8h*)(xn + (size_t)row * Dd + t * 8) = h;
}

// =====================================================================
// 2) f32 -> f16 weight cast (4 elems/thread)
// =====================================================================
__global__ __launch_bounds__(256) void cast_f32_f16_kernel(
    const float* __restrict__ in, _Float16* __restrict__ out, int n)
{
    int i = (blockIdx.x * 256 + threadIdx.x) * 4;
    if (i + 3 < n) {
        float4 v = *(const float4*)(in + i);
        out[i + 0] = (_Float16)v.x;
        out[i + 1] = (_Float16)v.y;
        out[i + 2] = (_Float16)v.z;
        out[i + 3] = (_Float16)v.w;
    }
}

// =====================================================================
// 3) WMMA GEMM:  C[M,N] = A[M,K](f16) * W[N,K]^T(f16), f32 accumulate.
//    Block tile 128x128, Kblk=32, 256 threads = 8 waves (4x2 wave grid),
//    each wave 2x4 fragments of v_wmma_f32_16x16x32_f16.
//    - N, K are template constants: C-store offsets fold into the 24-bit
//      instruction offset (no per-store 64-bit VALU address chains).
//    - LDS double-buffered (ping-pong): next tile's global_load_b128s
//      issue before the current tile's ds gathers + WMMAs; one barrier
//      per K-step instead of two.
//    - LDS row stride 40 f16 (80B): the 16 b128 fragment gathers of a
//      wave hit 16 disjoint 4-bank groups -> conflict free.
// =====================================================================
template<int N, int K>
__global__ __launch_bounds__(256) void wmma_gemm_kernel(
    const _Float16* __restrict__ A,   // [M,K] row-major
    const _Float16* __restrict__ W,   // [N,K] row-major
    float* __restrict__ C)            // [M,N]
{
    __shared__ _Float16 As[2][128 * 40];
    __shared__ _Float16 Bs[2][128 * 40];

    const int t     = threadIdx.x;
    const int lane  = t & 31;
    const int wave  = t >> 5;
    const int wm    = wave & 3;     // 32-row slab
    const int wn    = wave >> 2;    // 64-col slab
    const int hi    = lane >> 4;    // 0/1 : K-half select
    const int r     = lane & 15;    // row (A) / col (B) within 16

    const int m0 = blockIdx.x * 128;
    const int n0 = blockIdx.y * 128;

    const int ldrow = t >> 1;             // 0..127
    const int ldcol = (t & 1) * 16;       // 0 / 16

    const _Float16* ag = A + (size_t)(m0 + ldrow) * K + ldcol;
    const _Float16* bg = W + (size_t)(n0 + ldrow) * K + ldcol;

    v8f acc[2][4] = {};

    // prologue: stage K-tile 0 into buffer 0
    {
        _Float16* as = &As[0][ldrow * 40 + ldcol];
        *(float4*)as       = *(const float4*)ag;
        *(float4*)(as + 8) = *(const float4*)(ag + 8);
        _Float16* bs = &Bs[0][ldrow * 40 + ldcol];
        *(float4*)bs       = *(const float4*)bg;
        *(float4*)(bs + 8) = *(const float4*)(bg + 8);
    }
    __syncthreads();

    const int nkt = K / 32;
    for (int kt = 0; kt < nkt; ++kt) {
        const int cur  = kt & 1;
        const bool more = (kt + 1) < nkt;

        // issue next tile's global loads early (latency overlapped w/ WMMA)
        float4 a0, a1, b0, b1;
        if (more) {
            const _Float16* an = ag + (kt + 1) * 32;
            a0 = *(const float4*)an;  a1 = *(const float4*)(an + 8);
            const _Float16* bn = bg + (kt + 1) * 32;
            b0 = *(const float4*)bn;  b1 = *(const float4*)(bn + 8);
        }

        // gather fragments from current buffer
        v16h afrag[2], bfrag[4];
        #pragma unroll
        for (int i = 0; i < 2; ++i) {
            const _Float16* p = &As[cur][(wm * 32 + i * 16 + r) * 40 + hi * 8];
            union { v16h h; float4 f[2]; } u;
            u.f[0] = *(const float4*)p;          // K 0..7   (or 8..15)
            u.f[1] = *(const float4*)(p + 16);   // K 16..23 (or 24..31)
            afrag[i] = u.h;
        }
        #pragma unroll
        for (int j = 0; j < 4; ++j) {
            const _Float16* p = &Bs[cur][(wn * 64 + j * 16 + r) * 40 + hi * 8];
            union { v16h h; float4 f[2]; } u;
            u.f[0] = *(const float4*)p;
            u.f[1] = *(const float4*)(p + 16);
            bfrag[j] = u.h;
        }
        #pragma unroll
        for (int i = 0; i < 2; ++i)
            #pragma unroll
            for (int j = 0; j < 4; ++j)
                acc[i][j] = __builtin_amdgcn_wmma_f32_16x16x32_f16(
                    false, afrag[i], false, bfrag[j],
                    (short)0, acc[i][j], false, false);

        // commit next tile to the other buffer (no conflict with readers of cur)
        if (more) {
            _Float16* as = &As[1 - cur][ldrow * 40 + ldcol];
            *(float4*)as       = a0;  *(float4*)(as + 8) = a1;
            _Float16* bs = &Bs[1 - cur][ldrow * 40 + ldcol];
            *(float4*)bs       = b0;  *(float4*)(bs + 8) = b1;
        }
        __syncthreads();
    }

    // C/D layout: lane = (m/8)*16 + n, vgpr = m%8
    #pragma unroll
    for (int i = 0; i < 2; ++i) {
        float* cp = C + (size_t)(m0 + wm * 32 + i * 16 + hi * 8) * N
                      + n0 + wn * 64 + r;
        #pragma unroll
        for (int j = 0; j < 4; ++j)
            #pragma unroll
            for (int v = 0; v < 8; ++v)
                cp[(size_t)v * N + j * 16] = acc[i][j][v];   // immediate offsets
    }
}

// =====================================================================
// 4) gk path: xr = xn @ w1^T (rank 16), g = logsigmoid(xr @ w2^T + b)/16
//    One block per token row.
// =====================================================================
__global__ __launch_bounds__(256) void gk_kernel(
    const _Float16* __restrict__ xn,   // [ROWS, D]
    const float* __restrict__ w1,      // [16, D]
    const float* __restrict__ w2,      // [NK, 16]
    const float* __restrict__ b2,      // [NK]
    float* __restrict__ g)             // [ROWS, NK]
{
    __shared__ float part[256];
    __shared__ float xr[16];
    const int row  = blockIdx.x;
    const int t    = threadIdx.x;
    const int rank = t >> 4;    // 0..15
    const int seg  = t & 15;    // 128 elems each

    const v8h*    xp = (const v8h*)   (xn + (size_t)row  * Dd + seg * 128);
    const float4* wp = (const float4*)(w1 + (size_t)rank * Dd + seg * 128);
    float s = 0.f;
    #pragma unroll
    for (int i = 0; i < 16; ++i) {           // 16 x 8 = 128 elems
        v8h    xv = xp[i];
        float4 w0 = wp[2 * i];
        float4 w1v = wp[2 * i + 1];
        s += (float)xv[0] * w0.x + (float)xv[1] * w0.y
           + (float)xv[2] * w0.z + (float)xv[3] * w0.w
           + (float)xv[4] * w1v.x + (float)xv[5] * w1v.y
           + (float)xv[6] * w1v.z + (float)xv[7] * w1v.w;
    }
    part[t] = s;
    __syncthreads();
    if (seg == 0) {
        float tot = 0.f;
        #pragma unroll
        for (int i = 0; i < 16; ++i) tot += part[rank * 16 + i];
        xr[rank] = tot;
    }
    __syncthreads();

    #pragma unroll
    for (int o = 0; o < 4; ++o) {
        const int n = t * 4 + o;
        float acc = b2[n];
        #pragma unroll
        for (int i = 0; i < 16; ++i) acc += xr[i] * w2[n * 16 + i];
        // log_sigmoid(x) = min(x,0) - log1p(exp(-|x|))
        const float ls = fminf(acc, 0.f) - log1pf(expf(-fabsf(acc)));
        g[(size_t)row * NK + n] = ls * (1.0f / 16.0f);
    }
}

// =====================================================================
// 5) Chunked GLA recurrence.  One block per (b,h); state 64x128 f32 held
//    in registers (32/thread); k, exp(g), v chunks staged in 64KB LDS.
// =====================================================================
__global__ __launch_bounds__(256) void gla_scan_kernel(
    const float* __restrict__ kbuf,  // [ROWS, NK]
    const float* __restrict__ vbuf,  // [ROWS, NV]
    const float* __restrict__ gbuf,  // [ROWS, NK]
    const float* __restrict__ S0,    // [B,H,DK,DV]
    float* __restrict__ out)         // states [B,32,H,DK,DV] ++ final [B,H,DK,DV]
{
    __shared__ float kls[CHUNK * DK];    // 16 KB
    __shared__ float egs[CHUNK * DK];    // 16 KB
    __shared__ float vls[CHUNK * DV];    // 32 KB

    const int blk = blockIdx.x;          // 0..31
    const int b   = blk >> 4;
    const int h   = blk & 15;
    const int t   = threadIdx.x;
    const int dv  = t & 127;
    const int dk0 = (t >> 7) * 32;

    float s[32];
    const float* s0 = S0 + ((size_t)(b * Hh + h)) * DK * DV;
    #pragma unroll
    for (int i = 0; i < 32; ++i) s[i] = s0[(dk0 + i) * DV + dv];

    for (int c = 0; c < NCHUNK; ++c) {
        #pragma unroll
        for (int rep = 0; rep < 16; ++rep) {           // 64x64 k and exp(g)
            const int idx = t + rep * 256;
            const int j = idx >> 6, dk = idx & 63;
            const size_t m = (size_t)b * Tt + c * CHUNK + j;
            kls[idx] = kbuf[m * NK + h * DK + dk];
            egs[idx] = expf(gbuf[m * NK + h * DK + dk]);
        }
        #pragma unroll
        for (int rep = 0; rep < 32; ++rep) {           // 64x128 v
            const int idx = t + rep * 256;
            const int j = idx >> 7, dvv = idx & 127;
            const size_t m = (size_t)b * Tt + c * CHUNK + j;
            vls[idx] = vbuf[m * NV + h * DV + dvv];
        }
        __syncthreads();

        for (int j = 0; j < CHUNK; ++j) {
            const float vv = vls[j * DV + dv];
            const float* kp = &kls[j * DK + dk0];
            const float* ep = &egs[j * DK + dk0];
            #pragma unroll
            for (int i = 0; i < 32; ++i)
                s[i] = fmaf(kp[i], vv, s[i] * ep[i]);
        }
        __syncthreads();

        float* op = out + (((size_t)b * NCHUNK + c) * Hh + h) * DK * DV;
        #pragma unroll
        for (int i = 0; i < 32; ++i) op[(dk0 + i) * DV + dv] = s[i];
    }
    float* fp = out + (size_t)Bb * NCHUNK * Hh * DK * DV
                    + ((size_t)(b * Hh + h)) * DK * DV;
    #pragma unroll
    for (int i = 0; i < 32; ++i) fp[(dk0 + i) * DV + dv] = s[i];
}

// =====================================================================
// Host orchestration
// =====================================================================
extern "C" void kernel_launch(void* const* d_in, const int* in_sizes, int n_in,
                              void* d_out, int out_size, void* d_ws, size_t ws_size,
                              hipStream_t stream) {
    const float* hidden = (const float*)d_in[0];
    const float* S0     = (const float*)d_in[1];
    const float* nw     = (const float*)d_in[2];
    const float* k_w    = (const float*)d_in[3];
    const float* v_w    = (const float*)d_in[4];
    const float* gk_w1  = (const float*)d_in[5];
    const float* gk_w2  = (const float*)d_in[6];
    const float* gk_b2  = (const float*)d_in[7];
    float* out = (float*)d_out;

    char* ws = (char*)d_ws;
    _Float16* xn   = (_Float16*)ws;  ws += (size_t)ROWS * Dd * sizeof(_Float16);
    _Float16* kw16 = (_Float16*)ws;  ws += (size_t)NK * Dd * sizeof(_Float16);
    _Float16* vw16 = (_Float16*)ws;  ws += (size_t)NV * Dd * sizeof(_Float16);
    float* kb = (float*)ws;          ws += (size_t)ROWS * NK * sizeof(float);
    float* vb = (float*)ws;          ws += (size_t)ROWS * NV * sizeof(float);
    float* gb = (float*)ws;          ws += (size_t)ROWS * NK * sizeof(float);

    // 1) RMSNorm + cast activations
    rmsnorm_cast_kernel<<<ROWS, 256, 0, stream>>>(hidden, nw, xn);
    // 2) cast weights
    {
        int nk = NK * Dd, nv = NV * Dd;
        cast_f32_f16_kernel<<<(nk / 4 + 255) / 256, 256, 0, stream>>>(k_w, kw16, nk);
        cast_f32_f16_kernel<<<(nv / 4 + 255) / 256, 256, 0, stream>>>(v_w, vw16, nv);
    }
    // 3) projections via WMMA (N,K compile-time)
    wmma_gemm_kernel<NK, Dd><<<dim3(ROWS / 128, NK / 128), 256, 0, stream>>>(
        xn, kw16, kb);
    wmma_gemm_kernel<NV, Dd><<<dim3(ROWS / 128, NV / 128), 256, 0, stream>>>(
        xn, vw16, vb);
    // 4) gate path
    gk_kernel<<<ROWS, 256, 0, stream>>>(xn, gk_w1, gk_w2, gk_b2, gb);
    // 5) sequential state recurrence + chunk state emission
    gla_scan_kernel<<<Bb * Hh, 256, 0, stream>>>(kb, vb, gb, S0, out);
}
